// LSTM_19748259627063
// MI455X (gfx1250) — matrix operand order repaired and checked by reading, not compiled, not run on hardware
//
#include <hip/hip_runtime.h>

typedef __attribute__((ext_vector_type(16))) __bf16 bf16x16;
typedef __attribute__((ext_vector_type(8)))  __bf16 bf16x8;
typedef __attribute__((ext_vector_type(8)))  float  f32x8;

#define T_SEQ 4096
#define IN_SZ 512
#define HID1  1024
#define HID2  2048
#define OUT_SZ 512

__device__ __forceinline__ unsigned short f2bf_rne(float x) {
  unsigned u = __float_as_uint(x);
  u += 0x7FFFu + ((u >> 16) & 1u);
  return (unsigned short)(u >> 16);
}

__device__ __forceinline__ float sigmoidf_(float x) {
  return 1.0f / (1.0f + __expf(-x));
}

// Async global->LDS copy of one 16B chunk (per-lane addresses), ASYNCcnt-tracked.
__device__ __forceinline__ void async_copy_b128(unsigned lds_addr,
                                                const void* gaddr) {
  asm volatile("global_load_async_to_lds_b128 %0, %1, off"
               :: "v"(lds_addr), "v"((unsigned long long)gaddr)
               : "memory");
}

// ---------------------------------------------------------------- utilities
__global__ void zero_u32(unsigned* p, int n) {
  int i = blockIdx.x * blockDim.x + threadIdx.x;
  if (i < n) p[i] = 0u;
}

__global__ void cvt_f32_bf16(const float* __restrict__ src,
                             unsigned short* __restrict__ dst, int n) {
  for (int i = blockIdx.x * blockDim.x + threadIdx.x; i < n;
       i += gridDim.x * blockDim.x)
    dst[i] = f2bf_rne(src[i]);
}

// --------------------------------------------------- WMMA GEMM with bias
// C[M,N] = A_bf[M,K] @ W_bf[N,K]^T + b0[N] + b1[N]
// block = 256 threads (8 waves); WG tile = 128(M) x 64(N).
// B (weight) tile is shared by all 8 waves -> staged in LDS via async DMA,
// double-buffered across 32-wide K chunks. A is read directly per-lane.
__global__ void wmma_gemm_bias(const unsigned short* __restrict__ A,
                               const unsigned short* __restrict__ W,
                               const float* __restrict__ b0,
                               const float* __restrict__ b1,
                               float* __restrict__ C,
                               int M, int N, int K) {
  __shared__ unsigned short Bs[2][64 * 32];      // 2 x 4KB double buffer
  const int tid  = threadIdx.x;
  const int lane = tid & 31;
  const int wave = tid >> 5;
  const int mrow = lane & 15;
  const int ksel = lane >> 4;
  const int m0 = (blockIdx.y * 8 + wave) * 16;
  const int n0 = blockIdx.x * 64;

  // async-copy assignment: thread -> (row n = tid>>2, 8-elem quarter = tid&3)
  const int cn  = tid >> 2;
  const int ckq = tid & 3;
  const unsigned short* wsrc = W + (size_t)(n0 + cn) * K + ckq * 8;
  const unsigned lds0 = (unsigned)(size_t)&Bs[0][cn * 32 + ckq * 8];
  const unsigned lds1 = (unsigned)(size_t)&Bs[1][cn * 32 + ckq * 8];

  const unsigned short* arow = A + (size_t)(m0 + mrow) * K;

  f32x8 acc0 = {}, acc1 = {}, acc2 = {}, acc3 = {};
  const int nch = K / 32;
  async_copy_b128(lds0, wsrc);                   // prologue: chunk 0

  for (int kc = 0; kc < nch; ++kc) {
    const int k0 = kc * 32;
    if (kc + 1 < nch) {
      async_copy_b128((kc & 1) ? lds0 : lds1, wsrc + k0 + 32);
      asm volatile("s_wait_asynccnt 0x1" ::: "memory");
    } else {
      asm volatile("s_wait_asynccnt 0x0" ::: "memory");
    }
    __syncthreads();                             // buffer kc&1 is ready

    // A fragment: 16x32 bf16, lane holds 16 contiguous k (k = 16*ksel + s)
    bf16x8 a0 = *(const bf16x8*)(arow + k0 + ksel * 16);
    bf16x8 a1 = *(const bf16x8*)(arow + k0 + 8 + ksel * 16);
    __builtin_prefetch((const void*)(arow + k0 + 128), 0, 3);
    bf16x16 Af;
#pragma unroll
    for (int i = 0; i < 8; ++i) { Af[i] = a0[i]; Af[i + 8] = a1[i]; }

    const unsigned short* bb = &Bs[kc & 1][0];
    bf16x16 B0 = *(const bf16x16*)(bb + (( 0 + mrow) * 32 + ksel * 16));
    bf16x16 B1 = *(const bf16x16*)(bb + ((16 + mrow) * 32 + ksel * 16));
    bf16x16 B2 = *(const bf16x16*)(bb + ((32 + mrow) * 32 + ksel * 16));
    bf16x16 B3 = *(const bf16x16*)(bb + ((48 + mrow) * 32 + ksel * 16));

    acc0 = __builtin_amdgcn_wmma_f32_16x16x32_bf16(false, Af, false, B0,
                                                   (short)0, acc0, false, false);
    acc1 = __builtin_amdgcn_wmma_f32_16x16x32_bf16(false, Af, false, B1,
                                                   (short)0, acc1, false, false);
    acc2 = __builtin_amdgcn_wmma_f32_16x16x32_bf16(false, Af, false, B2,
                                                   (short)0, acc2, false, false);
    acc3 = __builtin_amdgcn_wmma_f32_16x16x32_bf16(false, Af, false, B3,
                                                   (short)0, acc3, false, false);
    __syncthreads();                // all waves done reading before refill
  }

#pragma unroll
  for (int j = 0; j < 8; ++j) {
    const int row = m0 + j + 8 * ksel;
    float* cr = C + (size_t)row * N + n0;
    const int c0 = n0 + mrow;
    cr[mrow +  0] = acc0[j] + b0[c0 +  0] + b1[c0 +  0];
    cr[mrow + 16] = acc1[j] + b0[c0 + 16] + b1[c0 + 16];
    cr[mrow + 32] = acc2[j] + b0[c0 + 32] + b1[c0 + 32];
    cr[mrow + 48] = acc3[j] + b0[c0 + 48] + b1[c0 + 48];
  }
}

// ------------------------------------------------------- grid-wide barrier
__device__ __forceinline__ void grid_barrier(unsigned* bar, unsigned nwg) {
  __syncthreads();
  if (threadIdx.x == 0) {
    unsigned gen = __hip_atomic_load(&bar[1], __ATOMIC_ACQUIRE,
                                     __HIP_MEMORY_SCOPE_AGENT);
    unsigned prev = __hip_atomic_fetch_add(&bar[0], 1u, __ATOMIC_ACQ_REL,
                                           __HIP_MEMORY_SCOPE_AGENT);
    if (prev == nwg - 1u) {
      __hip_atomic_store(&bar[0], 0u, __ATOMIC_RELAXED,
                         __HIP_MEMORY_SCOPE_AGENT);
      __hip_atomic_fetch_add(&bar[1], 1u, __ATOMIC_RELEASE,
                             __HIP_MEMORY_SCOPE_AGENT);
    } else {
      while (__hip_atomic_load(&bar[1], __ATOMIC_ACQUIRE,
                               __HIP_MEMORY_SCOPE_AGENT) == gen)
        __builtin_amdgcn_s_sleep(2);
    }
  }
  __syncthreads();
}

// ------------------------------------------------- persistent LSTM recurrence
// Weights for this WG's gate-row slice are made LDS-RESIDENT (async DMA once),
// so each of the 4096 sequential steps reads weights at WGP-local bandwidth.
//   HD: hidden size.  DS: hidden units per WG (slice).  KS: K-split per chunk.
//   LDS weight slice = 4*DS*HD bf16 = 256KB for both instantiations.
//   8 waves = (DS/4 chunks of 16 gate rows) x KS K-halves.
// h_hist: bf16 [(T+1), HD], row 0 pre-zeroed; step t reads row t, writes t+1.
template <int HD, int DS, int KS>
__global__ void lstm_recur_lds(const float* __restrict__ pre,   // [T, 4*HD]
                               const unsigned short* __restrict__ whh, // [4*HD, HD]
                               unsigned short* __restrict__ h_hist,
                               float* __restrict__ c,
                               float* __restrict__ h_f32,
                               int nsteps, unsigned* bar) {
  constexpr int ROWS = 4 * DS;                    // gate rows per WG
  __shared__ unsigned short wlds[ROWS * HD];      // 256 KB weight slice
  __shared__ float gs[KS][ROWS];

  const int tid  = threadIdx.x;
  const int lane = tid & 31;
  const int wave = tid >> 5;
  const int mrow = lane & 15;
  const int ksel = lane >> 4;
  const int d0 = blockIdx.x * DS;
  const unsigned nwg = gridDim.x;

  // ---- one-time: DMA this WG's recurrent-weight slice into LDS
  constexpr int NCHK = ROWS * HD / 8;             // 16-byte chunks
  for (int i = tid; i < NCHK; i += 256) {
    const int lr = i / (HD / 8);
    const int k8 = (i % (HD / 8)) * 8;
    const int grow = (lr / DS) * HD + d0 + (lr % DS);   // global gate row
    async_copy_b128((unsigned)(size_t)&wlds[lr * HD + k8],
                    (const void*)(whh + (size_t)grow * HD + k8));
  }
  asm volatile("s_wait_asynccnt 0x0" ::: "memory");
  __syncthreads();

  const int ch    = wave / KS;                    // 16-row chunk id
  const int khalf = wave % KS;                    // K-split id
  const int grp   = ch / (DS / 16);               // gate group (i,f,g,o)
  const int sub   = ch % (DS / 16);
  const int lrbase = grp * DS + sub * 16 + mrow;  // LDS row for this lane
  constexpr int KW = HD / KS;
  const int kstart = khalf * KW;
  const unsigned short* wlr = &wlds[lrbase * HD];

  for (int t = 0; t < nsteps; ++t) {
    const unsigned short* hprev = h_hist + (size_t)t * HD;

    // ---- WMMA-GEMV: A = weight rows (LDS), B = broadcast h (global/L2).
    // Two accumulators halve the serial D->C dependency chain.
    f32x8 accE = {}, accO = {};
    for (int k0 = kstart; k0 < kstart + KW; k0 += 64) {
      {
        bf16x8 a0 = *(const bf16x8*)(wlr + k0 + ksel * 16);
        bf16x8 a1 = *(const bf16x8*)(wlr + k0 + 8 + ksel * 16);
        bf16x16 Af;
#pragma unroll
        for (int i = 0; i < 8; ++i) { Af[i] = a0[i]; Af[i + 8] = a1[i]; }
        bf16x16 Bf = *(const bf16x16*)(hprev + k0 + ksel * 16);
        accE = __builtin_amdgcn_wmma_f32_16x16x32_bf16(false, Af, false, Bf,
                                                       (short)0, accE, false, false);
      }
      {
        bf16x8 a0 = *(const bf16x8*)(wlr + k0 + 32 + ksel * 16);
        bf16x8 a1 = *(const bf16x8*)(wlr + k0 + 40 + ksel * 16);
        bf16x16 Af;
#pragma unroll
        for (int i = 0; i < 8; ++i) { Af[i] = a0[i]; Af[i + 8] = a1[i]; }
        bf16x16 Bf = *(const bf16x16*)(hprev + k0 + 32 + ksel * 16);
        accO = __builtin_amdgcn_wmma_f32_16x16x32_bf16(false, Af, false, Bf,
                                                       (short)0, accO, false, false);
      }
    }
    // D columns are all identical; lanes 0 and 16 publish rows j / j+8.
    if (mrow == 0) {
#pragma unroll
      for (int j = 0; j < 8; ++j)
        gs[khalf][grp * DS + sub * 16 + j + 8 * ksel] = accE[j] + accO[j];
    }
    __syncthreads();

    // ---- elementwise cell update for this slice
    if (tid < DS) {
      const int d = d0 + tid;
      const float* prow = pre + (size_t)t * 4 * HD;
      float s0 = gs[0][0 * DS + tid];
      float s1 = gs[0][1 * DS + tid];
      float s2 = gs[0][2 * DS + tid];
      float s3 = gs[0][3 * DS + tid];
      if constexpr (KS == 2) {
        s0 += gs[1][0 * DS + tid];
        s1 += gs[1][1 * DS + tid];
        s2 += gs[1][2 * DS + tid];
        s3 += gs[1][3 * DS + tid];
      }
      const float ig = sigmoidf_(s0 + prow[0 * HD + d]);
      const float fg = sigmoidf_(s1 + prow[1 * HD + d]);
      const float gg = tanhf    (s2 + prow[2 * HD + d]);
      const float og = sigmoidf_(s3 + prow[3 * HD + d]);
      const float cc = fg * c[d] + ig * gg;
      c[d] = cc;
      const float hh = og * tanhf(cc);
      h_f32[d] = hh;
      h_hist[(size_t)(t + 1) * HD + d] = f2bf_rne(hh);
    }
    __threadfence();
    grid_barrier(bar, nwg);
  }
}

// ---------------------------------------------------- small fp32 linear (GEMV)
__global__ void linear_f32(const float* __restrict__ x,
                           const float* __restrict__ W,  // [N, K]
                           const float* __restrict__ b,
                           float* __restrict__ y, int N, int K) {
  int n = blockIdx.x * blockDim.x + threadIdx.x;
  if (n >= N) return;
  const float* wr = W + (size_t)n * K;
  float s = b[n];
  for (int k = 0; k < K; ++k) s = fmaf(x[k], wr[k], s);
  y[n] = s;
}

// ---------------------------------------------------------------- launcher
extern "C" void kernel_launch(void* const* d_in, const int* in_sizes, int n_in,
                              void* d_out, int out_size, void* d_ws,
                              size_t ws_size, hipStream_t stream) {
  (void)in_sizes; (void)n_in; (void)out_size; (void)ws_size;
  const float* x    = (const float*)d_in[0];
  const float* wih1 = (const float*)d_in[1];
  const float* whh1 = (const float*)d_in[2];
  const float* bih1 = (const float*)d_in[3];
  const float* bhh1 = (const float*)d_in[4];
  const float* wih2 = (const float*)d_in[5];
  const float* whh2 = (const float*)d_in[6];
  const float* bih2 = (const float*)d_in[7];
  const float* bhh2 = (const float*)d_in[8];
  const float* wl1  = (const float*)d_in[9];
  const float* bl1  = (const float*)d_in[10];
  const float* wl2  = (const float*)d_in[11];
  const float* bl2  = (const float*)d_in[12];
  float* out = (float*)d_out;

  char* ws = (char*)d_ws;
  size_t off = 0;
  auto take = [&](size_t bytes) -> char* {
    char* p = ws + off;
    off += (bytes + 255) & ~(size_t)255;
    return p;
  };
  unsigned*       bar     = (unsigned*)take(256);
  unsigned short* x_bf    = (unsigned short*)take((size_t)T_SEQ * IN_SZ * 2);
  unsigned short* wih1_bf = (unsigned short*)take((size_t)4 * HID1 * IN_SZ * 2);
  unsigned short* whh1_bf = (unsigned short*)take((size_t)4 * HID1 * HID1 * 2);
  unsigned short* wih2_bf = (unsigned short*)take((size_t)4 * HID2 * HID1 * 2);
  unsigned short* whh2_bf = (unsigned short*)take((size_t)4 * HID2 * HID2 * 2);
  float*          pre1    = (float*)take((size_t)T_SEQ * 4 * HID1 * 4);
  float*          pre2    = (float*)take((size_t)T_SEQ * 4 * HID2 * 4);
  unsigned short* h1h     = (unsigned short*)take((size_t)(T_SEQ + 1) * HID1 * 2);
  unsigned short* h2h     = (unsigned short*)take((size_t)(T_SEQ + 1) * HID2 * 2);
  float*          c1      = (float*)take(HID1 * 4);
  float*          c2      = (float*)take(HID2 * 4);
  float*          h1f     = (float*)take(HID1 * 4);
  float*          h2f     = (float*)take(HID2 * 4);
  float*          p1      = (float*)take(2 * OUT_SZ * 4);

  // ---- state init (ws is not re-poisoned between replays; re-zero each call)
  zero_u32<<<1, 256, 0, stream>>>(bar, 64);
  zero_u32<<<(HID1 + 255) / 256, 256, 0, stream>>>((unsigned*)c1, HID1);
  zero_u32<<<(HID2 + 255) / 256, 256, 0, stream>>>((unsigned*)c2, HID2);
  zero_u32<<<(HID1 / 2 + 255) / 256, 256, 0, stream>>>((unsigned*)h1h, HID1 / 2);
  zero_u32<<<(HID2 / 2 + 255) / 256, 256, 0, stream>>>((unsigned*)h2h, HID2 / 2);

  // ---- fp32 -> bf16 conversions
  cvt_f32_bf16<<<1024, 256, 0, stream>>>(x,    x_bf,    T_SEQ * IN_SZ);
  cvt_f32_bf16<<<1024, 256, 0, stream>>>(wih1, wih1_bf, 4 * HID1 * IN_SZ);
  cvt_f32_bf16<<<2048, 256, 0, stream>>>(whh1, whh1_bf, 4 * HID1 * HID1);
  cvt_f32_bf16<<<2048, 256, 0, stream>>>(wih2, wih2_bf, 4 * HID2 * HID1);
  cvt_f32_bf16<<<4096, 256, 0, stream>>>(whh2, whh2_bf, 4 * HID2 * HID2);

  // ---- pre1 = x @ w_ih1^T + (b_ih1 + b_hh1)   [4096 x 4096]
  {
    dim3 g(4 * HID1 / 64, T_SEQ / 128);
    wmma_gemm_bias<<<g, 256, 0, stream>>>(x_bf, wih1_bf, bih1, bhh1, pre1,
                                          T_SEQ, 4 * HID1, IN_SZ);
  }
  // ---- LSTM layer 1 recurrence (persistent, 32 WGs, 256KB LDS weights)
  lstm_recur_lds<HID1, 32, 1><<<HID1 / 32, 256, 0, stream>>>(
      pre1, whh1_bf, h1h, c1, h1f, T_SEQ, bar);
  // ---- pre2 = h1 @ w_ih2^T + (b_ih2 + b_hh2)  [4096 x 8192]
  {
    dim3 g(4 * HID2 / 64, T_SEQ / 128);
    wmma_gemm_bias<<<g, 256, 0, stream>>>(h1h + HID1, wih2_bf, bih2, bhh2, pre2,
                                          T_SEQ, 4 * HID2, HID1);
  }
  // ---- LSTM layer 2 recurrence (persistent, 128 WGs, 256KB LDS weights)
  lstm_recur_lds<HID2, 16, 2><<<HID2 / 16, 256, 0, stream>>>(
      pre2, whh2_bf, h2h, c2, h2f, T_SEQ, bar);
  // ---- final linears on last hidden state only
  linear_f32<<<(2 * OUT_SZ + 255) / 256, 256, 0, stream>>>(h2f, wl1, bl1, p1,
                                                           2 * OUT_SZ, HID2);
  linear_f32<<<(OUT_SZ + 255) / 256, 256, 0, stream>>>(p1, wl2, bl2, out,
                                                       OUT_SZ, 2 * OUT_SZ);
}
